// GPMatchEncoderTRT_24644522344537
// MI455X (gfx1250) — matrix operand order repaired and checked by reading, not compiled, not run on hardware
//
#include <hip/hip_runtime.h>

typedef __bf16 bf16_t;
typedef __attribute__((ext_vector_type(16))) __bf16 v16bf;
typedef __attribute__((ext_vector_type(8)))  __bf16 v8bf;
typedef __attribute__((ext_vector_type(8)))  float  v8f;

#define BATCH 4
#define CH    256
#define NPIX  4096
#define BETA_F 14.285714285714286f
#define KEYS_PER_WAVE (NPIX / 4)   // 4 waves per block split the key range

// ---------------------------------------------------------------------------
// Kernel 1: per-(b,c) inverse L2 norm over the spatial dim (4096 elements).
// ---------------------------------------------------------------------------
__global__ __launch_bounds__(256) void rownorm_kernel(const float* __restrict__ src,
                                                      float* __restrict__ inv) {
  const int row = blockIdx.x;                       // 0 .. B*C-1
  const float* p = src + (size_t)row * NPIX;
  float s = 0.f;
  for (int i = threadIdx.x; i < NPIX; i += 256) {
    float v = p[i];
    s = fmaf(v, v, s);
  }
#pragma unroll
  for (int off = 16; off >= 1; off >>= 1)
    s += __shfl_xor(s, off, 32);
  __shared__ float red[8];
  if ((threadIdx.x & 31) == 0) red[threadIdx.x >> 5] = s;
  __syncthreads();
  if (threadIdx.x < 8) {
    s = red[threadIdx.x];
    s += __shfl_xor(s, 4, 32);
    s += __shfl_xor(s, 2, 32);
    s += __shfl_xor(s, 1, 32);
    if (threadIdx.x == 0) inv[row] = 1.0f / (sqrtf(s) + 1e-6f);
  }
}

// ---------------------------------------------------------------------------
// Kernel 2: normalize + convert to bf16 + transpose [B,C,N] -> [B,N,C].
// ---------------------------------------------------------------------------
__global__ __launch_bounds__(256) void norm_transpose_kernel(const float* __restrict__ src,
                                                             const float* __restrict__ inv,
                                                             bf16_t* __restrict__ dst) {
  __shared__ float tile[32][33];
  const int b  = blockIdx.z;
  const int n0 = blockIdx.x * 32;
  const int c0 = blockIdx.y * 32;
  const float* s  = src + (size_t)b * CH * NPIX;
  const float* iv = inv + b * CH;
#pragma unroll
  for (int i = threadIdx.y; i < 32; i += 8) {
    const int c = c0 + i;
    tile[i][threadIdx.x] = s[(size_t)c * NPIX + n0 + threadIdx.x] * iv[c];
  }
  __syncthreads();
  bf16_t* d = dst + (size_t)b * NPIX * CH;
#pragma unroll
  for (int i = threadIdx.y; i < 32; i += 8) {
    const int n = n0 + i;
    d[(size_t)n * CH + c0 + threadIdx.x] = (bf16_t)tile[threadIdx.x][i];
  }
}

// ---------------------------------------------------------------------------
// A-operand tile loader: 16 keys x 256 channels, CDNA5 16-bit A-matrix layout
// (lane = key M; elements 0..7 at K = half*8, 8..15 at K = 16 + half*8 within
// each 32-wide k-chunk). Fully unrolled -> static indexing -> registers.
// ---------------------------------------------------------------------------
__device__ __forceinline__ void load_a_tile(const bf16_t* __restrict__ brow,
                                            int half, v16bf* __restrict__ Af) {
#pragma unroll
  for (int k = 0; k < 8; ++k) {
    v8bf lo = *(const v8bf*)(brow + k * 32 + half * 8);
    v8bf hi = *(const v8bf*)(brow + k * 32 + 16 + half * 8);
    Af[k] = __builtin_shufflevector(lo, hi, 0, 1, 2, 3, 4, 5, 6, 7,
                                    8, 9, 10, 11, 12, 13, 14, 15);
  }
}

// ---------------------------------------------------------------------------
// Lane-local online softmax update over this lane's 8 keys of one 16-key tile
// (m = m0 + r + 8*half for accumulator element r).
// ---------------------------------------------------------------------------
__device__ __forceinline__ void online_update(const v8f& acc, int m0, int half,
                                              float& mx, float& Z,
                                              float& Sx, float& Sy) {
  float s[8];
  float tmax = -1e30f;
#pragma unroll
  for (int r = 0; r < 8; ++r) {
    s[r] = BETA_F * acc[r];
    tmax = fmaxf(tmax, s[r]);
  }
  const float nm   = fmaxf(mx, tmax);
  const float corr = __expf(mx - nm);
  Z *= corr; Sx *= corr; Sy *= corr;
  const float ybase = (float)(m0 >> 6);              // m/64 constant per tile
  const float xbase = (float)((m0 & 63) + half * 8); // m%64 base for this lane
  float esum = 0.f;
#pragma unroll
  for (int r = 0; r < 8; ++r) {
    const float e = __expf(s[r] - nm);
    esum += e;
    Sx = fmaf(e, xbase + (float)r, Sx);
  }
  Z += esum;
  Sy = fmaf(esum, ybase, Sy);
  mx = nm;
}

// ---------------------------------------------------------------------------
// Kernel 3: fused flash GEMM + softmax + coordinate expectation.
// One block (4 waves) per 16-query tile; each wave covers 1024 keys.
// Explicit ping-pong double buffering (A0/A1, static indexing only) so the
// 16 b128 loads of the next tile overlap the current tile's 8-WMMA chain.
// sim^T tiles: each lane's 8 accumulators belong to one query (N = lane%16),
// keys M = r + 8*half -> lane-local online softmax; shfl+LDS merges at end.
// ---------------------------------------------------------------------------
__global__ __launch_bounds__(128) void flash_kernel(const bf16_t* __restrict__ anorm,
                                                    const bf16_t* __restrict__ bnorm,
                                                    float* __restrict__ out) {
  const int lane = threadIdx.x & 31;
  const int wave = threadIdx.x >> 5;
  const int b    = blockIdx.x >> 8;
  const int n0   = (blockIdx.x & 255) << 4;
  const int half = lane >> 4;
  const int nl   = lane & 15;

  // Query fragments: B-operand (K x N), N = query = lane%16,
  // element e<8 -> K = k*32 + half*8 + e, e>=8 -> K = k*32 + 16 + half*8 + (e-8). Loop-invariant.
  const bf16_t* arow = anorm + ((size_t)b * NPIX + n0 + nl) * CH;
  v16bf Bfrag[8];
#pragma unroll
  for (int k = 0; k < 8; ++k) {
    v8bf lo = *(const v8bf*)(arow + k * 32 + half * 8);
    v8bf hi = *(const v8bf*)(arow + k * 32 + 16 + half * 8);
    Bfrag[k] = __builtin_shufflevector(lo, hi, 0, 1, 2, 3, 4, 5, 6, 7,
                                       8, 9, 10, 11, 12, 13, 14, 15);
  }

  const bf16_t* bbase = bnorm + (size_t)b * NPIX * CH + (size_t)nl * CH;

  const int m_begin = wave * KEYS_PER_WAVE;
  const int m_end   = m_begin + KEYS_PER_WAVE;   // multiple of 32 tiles

  float mx = -1e30f, Z = 0.f, Sx = 0.f, Sy = 0.f;

  v16bf A0[8], A1[8];
  load_a_tile(bbase + (size_t)m_begin * CH, half, A0);

#pragma unroll 1
  for (int m0 = m_begin; m0 < m_end; m0 += 32) {
    // Prefetch tile m0+16 (always exists: KEYS_PER_WAVE % 32 == 0).
    load_a_tile(bbase + (size_t)(m0 + 16) * CH, half, A1);
    {
      v8f acc = {};
#pragma unroll
      for (int k = 0; k < 8; ++k)
        acc = __builtin_amdgcn_wmma_f32_16x16x32_bf16(false, A0[k], false,
                                                      Bfrag[k], (short)0, acc,
                                                      false, false);
      online_update(acc, m0, half, mx, Z, Sx, Sy);
    }
    // Prefetch tile m0+32 (skip on last iteration; wave-uniform branch).
    if (m0 + 32 < m_end)
      load_a_tile(bbase + (size_t)(m0 + 32) * CH, half, A0);
    {
      v8f acc = {};
#pragma unroll
      for (int k = 0; k < 8; ++k)
        acc = __builtin_amdgcn_wmma_f32_16x16x32_bf16(false, A1[k], false,
                                                      Bfrag[k], (short)0, acc,
                                                      false, false);
      online_update(acc, m0 + 16, half, mx, Z, Sx, Sy);
    }
  }

  // Merge the two key-halves held by lane pair (L, L^16) — same query.
  {
    const float mo  = __shfl_xor(mx, 16, 32);
    const float Zo  = __shfl_xor(Z,  16, 32);
    const float Sxo = __shfl_xor(Sx, 16, 32);
    const float Syo = __shfl_xor(Sy, 16, 32);
    const float M  = fmaxf(mx, mo);
    const float c0 = __expf(mx - M);
    const float c1 = __expf(mo - M);
    mx = M;
    Z  = Z  * c0 + Zo  * c1;
    Sx = Sx * c0 + Sxo * c1;
    Sy = Sy * c0 + Syo * c1;
  }

  // Merge the 4 per-wave partial states through LDS.
  __shared__ float4 lds_state[4][16];
  if (half == 0) {
    float4 st; st.x = mx; st.y = Z; st.z = Sx; st.w = Sy;
    lds_state[wave][nl] = st;
  }
  __syncthreads();

  if (threadIdx.x < 16) {
    const int n = n0 + (int)threadIdx.x;
    float4 s0 = lds_state[0][threadIdx.x];
    float4 s1 = lds_state[1][threadIdx.x];
    float4 s2 = lds_state[2][threadIdx.x];
    float4 s3 = lds_state[3][threadIdx.x];
    const float M = fmaxf(fmaxf(s0.x, s1.x), fmaxf(s2.x, s3.x));
    const float c0 = __expf(s0.x - M);
    const float c1 = __expf(s1.x - M);
    const float c2 = __expf(s2.x - M);
    const float c3 = __expf(s3.x - M);
    const float Zt  = s0.y * c0 + s1.y * c1 + s2.y * c2 + s3.y * c3;
    const float Sxt = s0.z * c0 + s1.z * c1 + s2.z * c2 + s3.z * c3;
    const float Syt = s0.w * c0 + s1.w * c1 + s2.w * c2 + s3.w * c3;
    const float invZ = 1.0f / Zt;   // cert = exp(max - max)/Z = 1/Z
    out[(size_t)b * 2 * NPIX + n]        = Sxt * invZ;            // warp x
    out[(size_t)b * 2 * NPIX + NPIX + n] = Syt * invZ;            // warp y
    out[(size_t)BATCH * 2 * NPIX + (size_t)b * NPIX + n] = invZ;  // cert
  }
}

// ---------------------------------------------------------------------------
extern "C" void kernel_launch(void* const* d_in, const int* in_sizes, int n_in,
                              void* d_out, int out_size, void* d_ws, size_t ws_size,
                              hipStream_t stream) {
  (void)in_sizes; (void)n_in; (void)out_size; (void)ws_size;
  const float* f0 = (const float*)d_in[0];
  const float* f1 = (const float*)d_in[1];
  float* out = (float*)d_out;

  // Workspace: inv_a[B*C] f32 | inv_b[B*C] f32 | anorm[B*N*C] bf16 | bnorm[B*N*C] bf16
  float*  inv_a = (float*)d_ws;
  float*  inv_b = inv_a + BATCH * CH;
  bf16_t* anorm = (bf16_t*)(inv_b + BATCH * CH);
  bf16_t* bnorm = anorm + (size_t)BATCH * NPIX * CH;

  rownorm_kernel<<<BATCH * CH, 256, 0, stream>>>(f0, inv_a);
  rownorm_kernel<<<BATCH * CH, 256, 0, stream>>>(f1, inv_b);

  dim3 tgrid(NPIX / 32, CH / 32, BATCH);
  dim3 tblk(32, 8);
  norm_transpose_kernel<<<tgrid, tblk, 0, stream>>>(f0, inv_a, anorm);
  norm_transpose_kernel<<<tgrid, tblk, 0, stream>>>(f1, inv_b, bnorm);

  flash_kernel<<<BATCH * (NPIX / 16), 128, 0, stream>>>(anorm, bnorm, out);
}
